// CSEv1_14637248545545
// MI455X (gfx1250) — matrix operand order repaired and checked by reading, not compiled, not run on hardware
//
#include <hip/hip_runtime.h>
#include <hip/hip_bf16.h>

#define BSZ 16
#define SEQ 1024
#define NV  512
#define DM  256
#define KCL 8
#define EPS_INV 20.0f   // 1/0.05

typedef __attribute__((ext_vector_type(16))) __bf16 v16bf;
typedef __attribute__((ext_vector_type(8)))  float  v8f;
typedef __attribute__((ext_vector_type(4)))  float  f32x4;
typedef unsigned int u32x4 __attribute__((ext_vector_type(4)));
typedef int          i32x4 __attribute__((ext_vector_type(4)));
typedef int          i32x8 __attribute__((ext_vector_type(8)));

union FragBF {
    v16bf v;
    unsigned short h[16];
};

__device__ __forceinline__ void split_bf16(float f, unsigned short& hi, unsigned short& lo) {
    unsigned ub = __float_as_uint(f);
    hi = (unsigned short)(ub >> 16);                       // truncate -> bf16 hi
    float fhi = __uint_as_float(((unsigned)hi) << 16);
    float r = f - fhi;                                     // residual
    lo = (unsigned short)(__float_as_uint(r) >> 16);
}

// TDM descriptor (ISA ch.8): 2D tensor [SEQ][NV] f32, tile 16(x) x 32(y),
// LDS rows padded 16 DWORDs + 4 DWORDs  ==  stride-20 conflict-free layout.
__device__ __forceinline__ void tdm_load_tile(const float* gsrc, unsigned lds_off) {
    const unsigned long long ga = (unsigned long long)(size_t)(const void*)gsrc;
    u32x4 g0;
    g0[0] = 1u;                                    // count=1, user mode
    g0[1] = lds_off;                               // lds_addr (byte offset)
    g0[2] = (unsigned)ga;                          // global_addr[31:0]
    g0[3] = (unsigned)((ga >> 32) & 0x01FFFFFFu)   // global_addr[56:32]
            | (2u << 30);                          // type=2 ("image")
    i32x8 g1;
    g1[0] = (int)((2u << 16)        // data_size = 4B
                | (1u << 20)        // pad_enable
                | (3u << 22)        // pad_interval: 16 DWORDs
                | (3u << 25));      // pad_amount:   4 DWORDs
    g1[1] = (int)((unsigned)NV << 16);    // tensor_dim0 = 512
    g1[2] = (int)((unsigned)SEQ << 16);   // tensor_dim1 = 1024
    g1[3] = (int)(16u << 16);             // tile_dim0 = 16
    g1[4] = 32;                           // tile_dim1 = 32
    g1[5] = NV;                           // tensor_dim0_stride = 512
    g1[6] = 0;
    g1[7] = 0;
    i32x4 gz4 = {0, 0, 0, 0};             // groups 2/3 unused (2D tensor)
    i32x8 gz8 = {0, 0, 0, 0, 0, 0, 0, 0};
    __builtin_amdgcn_tensor_load_to_lds(g0, g1, gz4, gz4, gz8, 0);
}

// ---------------------------------------------------------------------------
// Kernel 1: x_emb[b,v,d] = sum_s x[b,s,v] * W[d,s] + bias[d]
// One wave = one 16(v) x 16(d) tile; 4 waves/block share the A tile.
// A tiles are double-buffered TDM DMAs (issue chunk i+1, s_wait_tensorcnt 1,
// compute chunk i) so DMA latency overlaps the WMMA stream.
// 3xBF16-split WMMA accumulation in f32 for near-fp32 fidelity.
// ---------------------------------------------------------------------------
__global__ void __launch_bounds__(128)
k_gemm_wmma(const float* __restrict__ x, const float* __restrict__ W,
            const float* __restrict__ bias, float* __restrict__ xemb) {
    __shared__ float As[2][32 * 20];  // two [k=32][m=16] tiles, row stride 20

    const int t    = threadIdx.x;     // 0..127
    const int wave = t >> 5;          // 0..3
    const int lane = t & 31;

    const int bid = blockIdx.x;       // 2048 blocks: b(16) * vtile(32) * dgroup(4)
    const int dg  = bid & 3;
    const int vt  = (bid >> 2) & 31;
    const int b   = bid >> 7;

    const int v0 = vt * 16;
    const int d0 = (dg * 4 + wave) * 16;

    const int m    = lane & 15;
    const int hi16 = lane >> 4;       // 0 for lanes 0-15, 1 for lanes 16-31

    // B fragment base: W[(d0 + n) * SEQ + hi16*16 + k0 + j], j=0..15 contiguous
    const float* Wrow  = W + (size_t)(d0 + m) * SEQ + hi16 * 16;
    const float* xbase = x + (size_t)b * SEQ * NV + v0;

    v8f c = {};

    if (wave == 0)                    // prime buffer 0 (TDM ignores EXEC; 1 issue/wave)
        tdm_load_tile(xbase, (unsigned)(size_t)(void*)&As[0][0]);

    for (int i = 0; i < SEQ / 32; ++i) {
        const int k0 = i * 32;
        if (wave == 0) {
            if (i + 1 < SEQ / 32) {   // prefetch next chunk into the other buffer
                tdm_load_tile(xbase + (size_t)(k0 + 32) * NV,
                              (unsigned)(size_t)(void*)&As[(i + 1) & 1][0]);
                __builtin_amdgcn_s_wait_tensorcnt(1);  // current tile landed
            } else {
                __builtin_amdgcn_s_wait_tensorcnt(0);  // last tile landed
            }
        }
        __syncthreads();                       // publish LDS tile to all waves

        const float* Ab = &As[i & 1][0];

        // A fragment, contiguous K order matching B: K = j + 16*(lane>=16)
        FragBF ahi, alo;
#pragma unroll
        for (int j = 0; j < 16; ++j) {
            const int K = j + hi16 * 16;
            split_bf16(Ab[K * 20 + m], ahi.h[j], alo.h[j]);
        }

        // B fragment: 16 contiguous floats per lane (L2-resident W)
        FragBF bhi, blo;
        const float* wp = Wrow + k0;
        __builtin_prefetch(wp + 32, 0, 1);   // next K-chunk -> global_prefetch_b8
#pragma unroll
        for (int jj = 0; jj < 4; ++jj) {
            const f32x4 w4 = *(const f32x4*)(wp + jj * 4);
            split_bf16(w4.x, bhi.h[jj * 4 + 0], blo.h[jj * 4 + 0]);
            split_bf16(w4.y, bhi.h[jj * 4 + 1], blo.h[jj * 4 + 1]);
            split_bf16(w4.z, bhi.h[jj * 4 + 2], blo.h[jj * 4 + 2]);
            split_bf16(w4.w, bhi.h[jj * 4 + 3], blo.h[jj * 4 + 3]);
        }

        // D = A*B + C : hi*hi + hi*lo + lo*hi  (3xBF16 split)
        c = __builtin_amdgcn_wmma_f32_16x16x32_bf16(false, ahi.v, false, bhi.v,
                                                    (short)0, c, false, false);
        c = __builtin_amdgcn_wmma_f32_16x16x32_bf16(false, ahi.v, false, blo.v,
                                                    (short)0, c, false, false);
        c = __builtin_amdgcn_wmma_f32_16x16x32_bf16(false, alo.v, false, bhi.v,
                                                    (short)0, c, false, false);

        __syncthreads();   // all waves done reading As[i&1] before it is re-DMA'd
    }

    const float bval = bias[d0 + m];
    // C layout: VGPR i -> M = i + 8*hi16, N = lane&15
#pragma unroll
    for (int i = 0; i < 8; ++i) {
        const int vrow = v0 + i + hi16 * 8;
        xemb[((size_t)b * NV + vrow) * DM + d0 + m] = c[i] + bval;
    }
}

// ---------------------------------------------------------------------------
// Kernel 2: per (b,v) row: cosine sims vs 8 cluster rows, then
// p_k = exp(sim_k/eps) / sum_k exp(sim_k/eps). One wave per row.
// ---------------------------------------------------------------------------
__global__ void __launch_bounds__(256)
k_prob(const float* __restrict__ xemb, const float* __restrict__ ce,
       float* __restrict__ p) {
    const int wave = threadIdx.x >> 5;
    const int lane = threadIdx.x & 31;
    const int row  = blockIdx.x * 8 + wave;   // 8192 rows

    const float* xr = xemb + (size_t)row * DM;

    float sx = 0.f;
    float dot[KCL], sc[KCL];
#pragma unroll
    for (int k = 0; k < KCL; ++k) { dot[k] = 0.f; sc[k] = 0.f; }

#pragma unroll
    for (int tt = 0; tt < DM / 32; ++tt) {
        const int col = lane + 32 * tt;
        const float xv = xr[col];
        sx = fmaf(xv, xv, sx);
#pragma unroll
        for (int k = 0; k < KCL; ++k) {
            const float cv = ce[k * DM + col];
            dot[k] = fmaf(xv, cv, dot[k]);
            sc[k]  = fmaf(cv, cv, sc[k]);
        }
    }
    // wave32 butterfly reduction
#pragma unroll
    for (int off = 16; off > 0; off >>= 1) {
        sx += __shfl_xor(sx, off, 32);
#pragma unroll
        for (int k = 0; k < KCL; ++k) {
            dot[k] += __shfl_xor(dot[k], off, 32);
            sc[k]  += __shfl_xor(sc[k],  off, 32);
        }
    }
    const float nx = fmaxf(sqrtf(sx), 1e-12f);
    float q[KCL], s = 0.f;
#pragma unroll
    for (int k = 0; k < KCL; ++k) {
        const float nc   = fmaxf(sqrtf(sc[k]), 1e-12f);
        const float prob = dot[k] / (nx * nc);
        q[k] = __expf(prob * EPS_INV);
        s += q[k];
    }
    if (lane == 0) {
        const float inv = 1.f / s;
#pragma unroll
        for (int k = 0; k < KCL; ++k) p[row * KCL + k] = q[k] * inv;
    }
}

// ---------------------------------------------------------------------------
// Kernel 3: out[k,b,v,s] = (u[k,b,v] < p[b,v,k]) ? 1 : 0, broadcast over s.
// threefry2x32-20 in JAX split-counter layout; non-temporal streaming stores
// (268 MB write stream -> keep it out of L2).
// ---------------------------------------------------------------------------
__device__ __forceinline__ unsigned rotl32(unsigned v, int r) {
    return (v << r) | (v >> (32 - r));
}

__device__ __forceinline__ float tf2x32_uniform(unsigned i) {
    const unsigned k0 = 0u, k1 = 7u;                 // jax.random.key(7)
    const unsigned k2 = 0x1BD11BDAu ^ k0 ^ k1;
    const unsigned half = (unsigned)(KCL * BSZ * NV) / 2u;   // 32768
    unsigned c0, c1; int which;
    if (i < half) { c0 = i;        c1 = i + half; which = 0; }
    else          { c0 = i - half; c1 = i;        which = 1; }

    const unsigned ks[3] = { k0, k1, k2 };
    const int R[8] = { 13, 15, 26, 6, 17, 29, 16, 24 };
    unsigned x0 = c0 + ks[0], x1 = c1 + ks[1];
#pragma unroll
    for (int blk = 0; blk < 5; ++blk) {
        const int base = (blk & 1) ? 4 : 0;
#pragma unroll
        for (int qr = 0; qr < 4; ++qr) {
            x0 += x1;
            x1 = rotl32(x1, R[base + qr]);
            x1 ^= x0;
        }
        const int j = blk + 1;
        x0 += ks[j % 3];
        x1 += ks[(j + 1) % 3] + (unsigned)j;
    }
    const unsigned bits = which ? x1 : x0;
    return __uint_as_float((bits >> 9) | 0x3f800000u) - 1.0f;   // [0,1)
}

__global__ void __launch_bounds__(256)
k_out(const float* __restrict__ p, float* __restrict__ out) {
    const unsigned row = blockIdx.x;          // ((k*BS)+b)*NV + v, 65536 rows
    const unsigned k   = row >> 13;           // / (BS*NV)
    const unsigned bv  = row & 8191u;         // b*NV + v
    const float pv  = p[bv * KCL + k];
    const float u   = tf2x32_uniform(row);
    const float val = (u < pv) ? 1.0f : 0.0f;
    const f32x4 v4  = { val, val, val, val };
    f32x4* dst = (f32x4*)(out + (size_t)row * SEQ);
    __builtin_nontemporal_store(v4, dst + threadIdx.x);   // 256 thr * 4 = 1024 floats
}

// ---------------------------------------------------------------------------
extern "C" void kernel_launch(void* const* d_in, const int* in_sizes, int n_in,
                              void* d_out, int out_size, void* d_ws, size_t ws_size,
                              hipStream_t stream) {
    const float* x    = (const float*)d_in[0];   // [16,1024,512]
    const float* W    = (const float*)d_in[1];   // [256,1024]
    const float* bias = (const float*)d_in[2];   // [256]
    const float* ce   = (const float*)d_in[3];   // [8,256]

    float* xemb = (float*)d_ws;                              // 16*512*256 f32 = 8 MB
    float* p    = xemb + (size_t)BSZ * NV * DM;              // 8192*8 f32
    float* out  = (float*)d_out;                             // [8,16,512,1024]

    k_gemm_wmma<<<BSZ * (NV / 16) * (DM / 64), 128, 0, stream>>>(x, W, bias, xemb);
    k_prob<<<(BSZ * NV) / 8, 256, 0, stream>>>(xemb, ce, p);
    k_out<<<KCL * BSZ * NV, 256, 0, stream>>>(p, out);
}
